// Net_34488587387331
// MI455X (gfx1250) — matrix operand order repaired and checked by reading, not compiled, not run on hardware
//
#include <hip/hip_runtime.h>

typedef _Float16 f16_t;
typedef __attribute__((ext_vector_type(16))) _Float16 v16h;
typedef __attribute__((ext_vector_type(8)))  _Float16 v8h;
typedef __attribute__((ext_vector_type(2)))  _Float16 v2h;
typedef __attribute__((ext_vector_type(8)))  float    v8f;

#define NNODES 10000
#define FEAT0  256
#define KDIM   10000

// ---------------------------------------------------------------------------
// Pre-swizzle emb (f32 [K,256]) into B-fragment order (f16), zero-padded in K.
// Layout: half index = ((kb*16 + cb)*32 + lane)*16 + j
//   lane<16 : element (k = 32*kb + (j<8 ? j : 8+j),  c = 16*cb + lane)
//   lane>=16: element (k = 32*kb + (j<8 ? 8+j : 16+j), c = 16*cb + lane-16)
// A wave's fragment for col-block cb is then 32 contiguous bytes per lane.
// ---------------------------------------------------------------------------
__global__ void k_swz_emb(const float* __restrict__ emb, f16_t* __restrict__ bswz,
                          int K, int total) {
    int tid = blockIdx.x * blockDim.x + threadIdx.x;
    if (tid >= total) return;
    int j    = tid & 15;
    int lane = (tid >> 4) & 31;
    int cb   = (tid >> 9) & 15;
    int kb   = tid >> 13;
    int k = kb * 32 + ((lane >> 4) << 3) + ((j >> 3) << 3) + j;
    int c = (cb << 4) + (lane & 15);
    float v = (k < K) ? emb[(size_t)k * FEAT0 + c] : 0.0f;
    bswz[tid] = (f16_t)v;
}

// ---------------------------------------------------------------------------
// Big GEMM: H0 = relu(X[10000,10000] @ emb[10000,256]) in f16 out.
// Block = 256 threads (8 waves), covers 16 rows x all 256 cols -> X streamed
// exactly once (~17us memory roofline at 23.3 TB/s; f16 WMMA compute ~20us).
// Double-buffered LDS A-tile: ONE barrier per 32-wide K step. Tail (K%32=16)
// peeled so the 312-iteration hot loop is branch-free.
// ---------------------------------------------------------------------------
__global__ __launch_bounds__(256) void k_gemm1(const float* __restrict__ X,
                                               const f16_t* __restrict__ Bswz,
                                               f16_t* __restrict__ H0,
                                               int K) {
    __shared__ __align__(16) f16_t lA[2][16 * 32];

    const int t    = threadIdx.x;
    const int wave = t >> 5;
    const int lane = t & 31;
    const int m0   = blockIdx.x << 4;

    v8f acc0 = {};
    v8f acc1 = {};

    // A-fragment LDS read offsets (ISA 16-bit A 16x32 layout)
    const int row  = lane & 15;
    const int klo  = (lane >> 4) << 3;            // 0 or 8
    const int aoff = row * 32 + klo;

    // Cooperative A-tile load mapping: 512 f32 / 256 threads = 2 each
    const int idx2 = t << 1;
    const int lr = idx2 >> 5;                     // 0..15
    const int lk = idx2 & 31;                     // even
    const float* xrow = X + (size_t)(m0 + lr) * K + lk;

    // B fragment pointers (per-kb stride = 16*32*16 = 8192 halves)
    const f16_t* bf0 = Bswz + ((size_t)(2 * wave)     * 32 + lane) * 16;
    const f16_t* bf1 = Bswz + ((size_t)(2 * wave + 1) * 32 + lane) * 16;

    const int kfull = K >> 5;                     // 312 full tiles
    int kb = 0;
    for (; kb < kfull; ++kb) {
        // branch-free full-tile load + convert
        const float2 f2 = *(const float2*)(xrow + (kb << 5));
        __builtin_prefetch(xrow + (kb << 5) + 64, 0, 1);   // 2 tiles ahead

        f16_t* buf = lA[kb & 1];
        v2h pk; pk[0] = (f16_t)f2.x; pk[1] = (f16_t)f2.y;
        *(v2h*)&buf[lr * 32 + lk] = pk;            // ds_store_b32
        __syncthreads();                           // single barrier per step

        v8h alo = *(const v8h*)&buf[aoff];         // ds_load_b128
        v8h ahi = *(const v8h*)&buf[aoff + 16];    // ds_load_b128
        v16h af;
#pragma unroll
        for (int i = 0; i < 8; ++i) { af[i] = alo[i]; af[8 + i] = ahi[i]; }

        v16h b0 = *(const v16h*)(bf0 + (size_t)kb * 8192);
        v16h b1 = *(const v16h*)(bf1 + (size_t)kb * 8192);

        acc0 = __builtin_amdgcn_wmma_f32_16x16x32_f16(false, af, false, b0,
                                                      (short)0, acc0, false, false);
        acc1 = __builtin_amdgcn_wmma_f32_16x16x32_f16(false, af, false, b1,
                                                      (short)0, acc1, false, false);
    }

    // peeled tail: remaining K % 32 values, zero-padded (Bswz is pre-padded)
    const int krem = K & 31;
    if (krem) {
        const int kbase = kfull << 5;
        float a0 = (lk < krem)     ? xrow[kbase]     : 0.0f;
        float a1 = (lk + 1 < krem) ? xrow[kbase + 1] : 0.0f;

        f16_t* buf = lA[kb & 1];
        v2h pk; pk[0] = (f16_t)a0; pk[1] = (f16_t)a1;
        *(v2h*)&buf[lr * 32 + lk] = pk;
        __syncthreads();

        v8h alo = *(const v8h*)&buf[aoff];
        v8h ahi = *(const v8h*)&buf[aoff + 16];
        v16h af;
#pragma unroll
        for (int i = 0; i < 8; ++i) { af[i] = alo[i]; af[8 + i] = ahi[i]; }

        v16h b0 = *(const v16h*)(bf0 + (size_t)kb * 8192);
        v16h b1 = *(const v16h*)(bf1 + (size_t)kb * 8192);

        acc0 = __builtin_amdgcn_wmma_f32_16x16x32_f16(false, af, false, b0,
                                                      (short)0, acc0, false, false);
        acc1 = __builtin_amdgcn_wmma_f32_16x16x32_f16(false, af, false, b1,
                                                      (short)0, acc1, false, false);
    }

    // Epilogue: ISA C/D layout (lane<16 -> M=r, lane>=16 -> M=8+r; N=lane%16)
    const int nlo  = lane & 15;
    const int moff = (lane < 16) ? 0 : 8;
    f16_t* orow = H0 + (size_t)(m0 + moff) * FEAT0 + (wave << 5) + nlo;
#pragma unroll
    for (int r = 0; r < 8; ++r) {
        float v0 = acc0[r]; v0 = v0 > 0.f ? v0 : 0.f;
        float v1 = acc1[r]; v1 = v1 > 0.f ? v1 : 0.f;
        orow[(size_t)r * FEAT0]      = (f16_t)v0;
        orow[(size_t)r * FEAT0 + 16] = (f16_t)v1;
    }
}

// ---------------------------------------------------------------------------
// Degree / normalization (deg counts dst + self-loop; dinv = deg^-0.5)
// ---------------------------------------------------------------------------
__global__ void k_deg_init(float* deg, int n) {
    int i = blockIdx.x * blockDim.x + threadIdx.x;
    if (i < n) deg[i] = 1.0f;                      // self-loop
}
__global__ void k_deg_count(const int* __restrict__ dst, float* deg, int E) {
    int e = blockIdx.x * blockDim.x + threadIdx.x;
    if (e < E) atomicAdd(&deg[dst[e]], 1.0f);
}
__global__ void k_dinv(const float* __restrict__ deg, float* dinv, int n) {
    int i = blockIdx.x * blockDim.x + threadIdx.x;
    if (i < n) { float d = deg[i]; dinv[i] = d > 0.f ? rsqrtf(d) : 0.f; }
}

// ---------------------------------------------------------------------------
// Small dense linears (all operands L2-resident; negligible vs big GEMM)
// ---------------------------------------------------------------------------
template <int N>
__global__ void k_lin_h(const f16_t* __restrict__ A, const float* __restrict__ W,
                        float* __restrict__ out, int M, int K) {
    int idx = blockIdx.x * blockDim.x + threadIdx.x;
    if (idx >= M * N) return;
    int m = idx / N, n = idx % N;
    const f16_t* ar = A + (size_t)m * K;
    float s = 0.f;
    for (int k = 0; k < K; ++k) s += (float)ar[k] * W[(size_t)k * N + n];
    out[idx] = s;
}
template <int N>
__global__ void k_lin_f(const float* __restrict__ A, const float* __restrict__ W,
                        float* __restrict__ out, int M, int K) {
    int idx = blockIdx.x * blockDim.x + threadIdx.x;
    if (idx >= M * N) return;
    int m = idx / N, n = idx % N;
    const float* ar = A + (size_t)m * K;
    float s = 0.f;
    for (int k = 0; k < K; ++k) s += ar[k] * W[(size_t)k * N + n];
    out[idx] = s;
}

// ---------------------------------------------------------------------------
// Edge aggregation: acc[dst] += h[src] * dinv[src]*dinv[dst] (f32 atomics)
// ---------------------------------------------------------------------------
template <int F>
__global__ void k_scatter(const float* __restrict__ h, const int* __restrict__ src,
                          const int* __restrict__ dst, const float* __restrict__ dinv,
                          float* __restrict__ acc, int E) {
    const int t = threadIdx.x;
    const int epb = 256 / F;
    int e = blockIdx.x * epb + t / F;
    if (e >= E) return;
    int f = t % F;
    int s = src[e], d = dst[e];
    float nrm = dinv[s] * dinv[d];
    atomicAdd(&acc[(size_t)d * F + f], h[(size_t)s * F + f] * nrm);
}

// Finalize: out = relu(acc + self_loop_term + bias)
template <int F>
__global__ void k_fin(const float* __restrict__ acc, const float* __restrict__ tl,
                      const float* __restrict__ dinv, const float* __restrict__ b,
                      float* __restrict__ out, int M) {
    int idx = blockIdx.x * blockDim.x + threadIdx.x;
    if (idx >= M * F) return;
    int m = idx / F, f = idx % F;
    float di = dinv[m];
    float v = acc[idx] + tl[idx] * di * di + b[f];
    out[idx] = v > 0.f ? v : 0.f;
}

// ---------------------------------------------------------------------------
extern "C" void kernel_launch(void* const* d_in, const int* in_sizes, int n_in,
                              void* d_out, int out_size, void* d_ws, size_t ws_size,
                              hipStream_t stream) {
    (void)n_in; (void)out_size; (void)ws_size;
    const float* x    = (const float*)d_in[0];
    const int*   ei   = (const int*)d_in[1];     // [2, E] int32
    const float* emb  = (const float*)d_in[2];
    const float* W1   = (const float*)d_in[3];
    const float* b1   = (const float*)d_in[4];
    const float* W2   = (const float*)d_in[5];
    const float* b2   = (const float*)d_in[6];
    const float* dW   = (const float*)d_in[7];
    float* out = (float*)d_out;

    const int Nn = NNODES, K = KDIM;
    const int E  = in_sizes[1] / 2;
    const int KB = (K + 31) / 32;                // 313
    const int* src = ei;
    const int* dst = ei + E;

    // carve workspace (256B-aligned chunks)
    char* ws = (char*)d_ws;
    size_t off = 0;
    auto carve = [&](size_t bytes) -> char* {
        char* p = ws + off;
        off += (bytes + 255) & ~(size_t)255;
        return p;
    };
    f16_t* embswz = (f16_t*)carve((size_t)KB * 8192 * sizeof(f16_t));
    f16_t* h0     = (f16_t*)carve((size_t)Nn * FEAT0 * sizeof(f16_t));
    float* t1     = (float*)carve((size_t)Nn * 64 * sizeof(float));
    float* a1     = (float*)carve((size_t)Nn * 64 * sizeof(float));
    float* h1     = (float*)carve((size_t)Nn * 64 * sizeof(float));
    float* t2     = (float*)carve((size_t)Nn * 32 * sizeof(float));
    float* a2     = (float*)carve((size_t)Nn * 32 * sizeof(float));
    float* h2     = (float*)carve((size_t)Nn * 32 * sizeof(float));
    float* deg    = (float*)carve((size_t)Nn * sizeof(float));
    float* dinv   = (float*)carve((size_t)Nn * sizeof(float));

    const int B = 256;
    auto g = [](long long n, int b) { return (unsigned)((n + b - 1) / b); };

    // 1) swizzle emb -> f16 B-fragment layout
    int swzTotal = KB * 8192;
    k_swz_emb<<<g(swzTotal, B), B, 0, stream>>>(emb, embswz, K, swzTotal);

    // 2) degrees + dinv
    k_deg_init<<<g(Nn, B), B, 0, stream>>>(deg, Nn);
    k_deg_count<<<g(E, B), B, 0, stream>>>(dst, deg, E);
    k_dinv<<<g(Nn, B), B, 0, stream>>>(deg, dinv, Nn);

    // 3) H0 = relu(x @ emb)  (WMMA, x streamed once)
    k_gemm1<<<Nn / 16, B, 0, stream>>>(x, embswz, h0, K);

    // 4) conv1: t1 = H0 @ W1; scatter; h1 = relu(acc + self + b1)
    k_lin_h<64><<<g((long long)Nn * 64, B), B, 0, stream>>>(h0, W1, t1, Nn, FEAT0);
    hipMemsetAsync(a1, 0, (size_t)Nn * 64 * sizeof(float), stream);
    k_scatter<64><<<g(E, 256 / 64), B, 0, stream>>>(t1, src, dst, dinv, a1, E);
    k_fin<64><<<g((long long)Nn * 64, B), B, 0, stream>>>(a1, t1, dinv, b1, h1, Nn);

    // 5) conv2: t2 = h1 @ W2; scatter; h2 = relu(acc + self + b2)
    k_lin_f<32><<<g((long long)Nn * 32, B), B, 0, stream>>>(h1, W2, t2, Nn, 64);
    hipMemsetAsync(a2, 0, (size_t)Nn * 32 * sizeof(float), stream);
    k_scatter<32><<<g(E, 256 / 32), B, 0, stream>>>(t2, src, dst, dinv, a2, E);
    k_fin<32><<<g((long long)Nn * 32, B), B, 0, stream>>>(a2, t2, dinv, b2, h2, Nn);

    // 6) decoder: out = h2 @ dec_W
    k_lin_f<16><<<g((long long)Nn * 16, B), B, 0, stream>>>(h2, dW, out, Nn, 32);
}
